// MaskedAttention_652835029571
// MI455X (gfx1250) — matrix operand order repaired and checked by reading, not compiled, not run on hardware
//
#include <hip/hip_runtime.h>

typedef __bf16 bf16;
typedef __attribute__((ext_vector_type(8)))  __bf16 v8bf;
typedef __attribute__((ext_vector_type(16))) __bf16 v16bf;
typedef __attribute__((ext_vector_type(8)))  float  v8f;
typedef __attribute__((ext_vector_type(4))) unsigned int u32x4;
typedef __attribute__((ext_vector_type(8))) int          i32x8;
typedef __attribute__((ext_vector_type(4))) int          i32x4;

union AFrag { v16bf v; v8bf h[2]; };

#define S_LEN  2048
#define NHEAD  16
#define DHEAD  64
#define DMODEL 1024
#define MROWS  4096   // B * S

static __device__ __forceinline__ v8f wmma_bf16(v16bf a, v16bf b, v8f c) {
    // (neg_a, A, neg_b, B, c_mod, C, reuse_a, reuse_b)
    return __builtin_amdgcn_wmma_f32_16x16x32_bf16(false, a, false, b, (short)0, c,
                                                   false, false);
}

// Issue a 2-D TDM tile load (bf16 elements) per CDNA5 D# layout (ISA 08 §8.3/8.4).
// dims/strides in elements (data_size = 2B); OOB rows/cols are zero-filled by TDM.
static __device__ __forceinline__ void tdm_load_2d(unsigned lds_off, const bf16* gptr,
                                                   int tile_w, int tile_h,
                                                   int tensor_w, int tensor_h,
                                                   int stride_elems) {
    unsigned long long ga = (unsigned long long)(uintptr_t)gptr;
    u32x4 g0;
    g0[0] = 1u;                                    // count=1, user descriptor
    g0[1] = lds_off;                               // LDS byte address
    g0[2] = (unsigned)ga;                          // global_addr[31:0]
    g0[3] = (unsigned)(ga >> 32) | (2u << 30);     // global_addr[56:32] | type=2
    i32x8 g1;
    g1[0] = (int)(1u << 16);                       // data_size=1 (2 bytes)
    g1[1] = (int)((unsigned)tensor_w << 16);       // tensor_dim0[15:0]
    g1[2] = (int)(((unsigned)tensor_w >> 16) | ((unsigned)tensor_h << 16));
    g1[3] = (int)(((unsigned)tensor_h >> 16) | ((unsigned)tile_w << 16));
    g1[4] = (int)(unsigned)tile_h;                 // tile_dim1 (tile_dim2=0)
    g1[5] = (int)(unsigned)stride_elems;           // tensor_dim0_stride[31:0]
    g1[6] = 0;
    g1[7] = 0;
    i32x4 gz4 = {0, 0, 0, 0};
    i32x8 gz8 = {0, 0, 0, 0, 0, 0, 0, 0};
    // clang-23 / therock form: (g0, g1, g2, g3, g_extra, cpol)
    __builtin_amdgcn_tensor_load_to_lds(g0, g1, gz4, gz4, gz8, 0);
}

// ---------------------------------------------------------------- casts ----
__global__ void cast_f32_bf16(const float* __restrict__ in, bf16* __restrict__ out, int n) {
    int i = blockIdx.x * blockDim.x + threadIdx.x;
    if (i < n) out[i] = (bf16)in[i];
}

// W [K=1024, N=1024] fp32 -> WT [N, K] bf16 (out-coalesced)
__global__ void cast_transpose(const float* __restrict__ W, bf16* __restrict__ WT) {
    int i = blockIdx.x * blockDim.x + threadIdx.x;   // index into WT
    int n = i >> 10;
    int k = i & 1023;
    WT[i] = (bf16)W[k * DMODEL + n];
}

// ----------------------------------------------------------------- GEMM ----
// Y = X[M=4096,K=1024] @ W[K,N=1024] + bias, W passed pre-transposed WT[N,K].
// mode 0: store bf16 as [bh, s, dh]      (Q, K)
// mode 2: store bf16 as [bh, dh, s]      (V transposed)
// mode 3: store fp32 row-major [M, N]    (output projection)
__global__ __launch_bounds__(256) void gemm_bf16(const bf16* __restrict__ X,
                                                 const bf16* __restrict__ WT,
                                                 const float* __restrict__ bias,
                                                 void* __restrict__ out, int mode) {
    const int tid    = threadIdx.x;
    const int wave   = tid >> 5;
    const int lane   = tid & 31;
    const int lanelo = lane & 15;
    const int hi     = lane >> 4;              // 0 or 1
    const int alo    = hi ? 16 : 0;            // A-fragment K sub-offset
    const int blo    = hi ? 16 : 0;            // B-fragment K sub-offset

    const int m0 = blockIdx.y * 128 + wave * 16;
    const int n0 = blockIdx.x * 64;

    const bf16* xp = X + (size_t)(m0 + lanelo) * DMODEL + alo;
    const bf16* wp[4];
#pragma unroll
    for (int g = 0; g < 4; ++g)
        wp[g] = WT + (size_t)(n0 + g * 16 + lanelo) * DMODEL + blo;

    v8f acc[4] = {};

    // software pipeline: fragments for step k+1 load while step k's WMMAs run;
    // unroll x2 so the ping-pong copies resolve to register renaming
    AFrag a;
    v16bf b[4];
    a.h[0] = *(const v8bf*)(xp);
    a.h[1] = *(const v8bf*)(xp + 8);
#pragma unroll
    for (int g = 0; g < 4; ++g) b[g] = *(const v16bf*)(wp[g]);

#pragma unroll 2
    for (int kb = 0; kb < DMODEL; kb += 32) {
        const int kn = (kb + 32 < DMODEL) ? (kb + 32) : kb;   // tail: dummy reload
        AFrag an;
        v16bf bn[4];
        an.h[0] = *(const v8bf*)(xp + kn);
        an.h[1] = *(const v8bf*)(xp + kn + 8);
#pragma unroll
        for (int g = 0; g < 4; ++g) bn[g] = *(const v16bf*)(wp[g] + kn);
        __builtin_prefetch(xp + kb + 256, 0, 1);              // global_prefetch_b8
#pragma unroll
        for (int g = 0; g < 4; ++g) acc[g] = wmma_bf16(a.v, b[g], acc[g]);
        a = an;
#pragma unroll
        for (int g = 0; g < 4; ++g) b[g] = bn[g];
    }

#pragma unroll
    for (int g = 0; g < 4; ++g) {
        const int n  = n0 + g * 16 + lanelo;
        const float bv = bias[n];
#pragma unroll
        for (int v = 0; v < 8; ++v) {
            const int r  = m0 + v + 8 * hi;          // global row: b*S + s
            const float val = acc[g][v] + bv;
            if (mode == 3) {
                ((float*)out)[(size_t)r * DMODEL + n] = val;
            } else {
                const int b2 = r / S_LEN, s = r % S_LEN;
                const int h  = n / DHEAD, dh = n % DHEAD;
                const int bh = b2 * NHEAD + h;
                if (mode == 2)
                    ((bf16*)out)[((size_t)bh * DHEAD + dh) * S_LEN + s] = (bf16)val;
                else
                    ((bf16*)out)[((size_t)bh * S_LEN + s) * DHEAD + dh] = (bf16)val;
            }
        }
    }
}

// ------------------------------------------------------- flash attention ----
// One wave per (bh, 16-query tile). Q,K: [bh, s, dh] bf16; VT: [bh, dh, s] bf16.
// K/V tiles staged by the Tensor Data Mover into double-buffered LDS.
__global__ __launch_bounds__(32) void flash_attn(const bf16* __restrict__ Q,
                                                 const bf16* __restrict__ K,
                                                 const bf16* __restrict__ VT,
                                                 bf16* __restrict__ ctx) {
    __shared__ __align__(64) bf16 lds_k[2][32 * DHEAD];   // 2 x 4KB
    __shared__ __align__(64) bf16 lds_v[2][DHEAD * 32];   // 2 x 4KB
    __shared__ __align__(64) bf16 lds_p[16 * 32];         // 1KB

    const int lane   = threadIdx.x & 31;
    const int lanelo = lane & 15;
    const int hi     = lane >> 4;
    const int alo    = hi ? 16 : 0;
    const int klo    = hi ? 16 : 0;

    const int qi = blockIdx.x & 127;          // S/16 tiles
    const int bh = blockIdx.x >> 7;
    const int q0 = qi * 16;

    const bf16* Qb = Q  + (size_t)bh * S_LEN * DHEAD;
    const bf16* Kb = K  + (size_t)bh * S_LEN * DHEAD;
    const bf16* Vb = VT + (size_t)bh * DHEAD * S_LEN;

    const unsigned koff[2] = { (unsigned)(uintptr_t)&lds_k[0][0],
                               (unsigned)(uintptr_t)&lds_k[1][0] };
    const unsigned voff[2] = { (unsigned)(uintptr_t)&lds_v[0][0],
                               (unsigned)(uintptr_t)&lds_v[1][0] };

    // Q fragments: rows q0..q0+15, K-dim = Dh (two 32-wide steps)
    AFrag qf[2];
    {
        const bf16* qrow = Qb + (size_t)(q0 + lanelo) * DHEAD;
        qf[0].h[0] = *(const v8bf*)(qrow + alo);
        qf[0].h[1] = *(const v8bf*)(qrow + alo + 8);
        qf[1].h[0] = *(const v8bf*)(qrow + 32 + alo);
        qf[1].h[1] = *(const v8bf*)(qrow + 32 + alo + 8);
    }

    v8f acc[4] = {};
    float mrow[8], lrow[8];
#pragma unroll
    for (int v = 0; v < 8; ++v) { mrow[v] = -__builtin_inff(); lrow[v] = 0.f; }

    // TDM prologue: stage key-tile 0
    tdm_load_2d(koff[0], Kb, DHEAD, 32, DHEAD, S_LEN, DHEAD);
    tdm_load_2d(voff[0], Vb, 32, DHEAD, S_LEN, DHEAD, S_LEN);

    int buf = 0;
    for (int t0 = 0; t0 < q0 + 16; t0 += 32) {
        // prefetch next tile into the other buffer (clamped; harmless on last iter)
        int t0n = t0 + 32;
        if (t0n > S_LEN - 32) t0n = S_LEN - 32;
        tdm_load_2d(koff[buf ^ 1], Kb + (size_t)t0n * DHEAD, DHEAD, 32,
                    DHEAD, S_LEN - t0n, DHEAD);
        tdm_load_2d(voff[buf ^ 1], Vb + t0n, 32, DHEAD,
                    S_LEN - t0n, DHEAD, S_LEN);
        // oldest two tensor ops (current tile) must have landed
        __builtin_amdgcn_s_wait_tensorcnt(2);
        asm volatile("" ::: "memory");

        const bf16* lk = &lds_k[buf][0];
        const bf16* lv = &lds_v[buf][0];

        // ----- scores: Q(16x64) @ K^T(64x32) -> two 16x16 C fragments
        v8f c0 = {}, c1 = {};
#pragma unroll
        for (int kk = 0; kk < 2; ++kk) {
            v16bf b0 = *(const v16bf*)(lk + lanelo * DHEAD + kk * 32 + klo);
            v16bf b1 = *(const v16bf*)(lk + (16 + lanelo) * DHEAD + kk * 32 + klo);
            c0 = wmma_bf16(qf[kk].v, b0, c0);
            c1 = wmma_bf16(qf[kk].v, b1, c1);
        }

        // ----- causal mask + scale + online softmax (rows live across 16 lanes)
        float p0[8], p1[8], alpha[8];
        const int sq_base = q0 + 8 * hi;
        const int kt0 = t0 + lanelo;
        const int kt1 = kt0 + 16;
#pragma unroll
        for (int v = 0; v < 8; ++v) {
            const int sq = sq_base + v;
            float s0 = (kt0 <= sq) ? c0[v] * 0.125f : -__builtin_inff();
            float s1 = (kt1 <= sq) ? c1[v] * 0.125f : -__builtin_inff();
            float mx = fmaxf(s0, s1);
#pragma unroll
            for (int off = 1; off < 16; off <<= 1)
                mx = fmaxf(mx, __shfl_xor(mx, off, 32));
            const float mnew = fmaxf(mrow[v], mx);
            const float a  = __expf(mrow[v] - mnew);
            const float e0 = __expf(s0 - mnew);
            const float e1 = __expf(s1 - mnew);
            float sm = e0 + e1;
#pragma unroll
            for (int off = 1; off < 16; off <<= 1)
                sm += __shfl_xor(sm, off, 32);
            lrow[v]  = lrow[v] * a + sm;
            mrow[v]  = mnew;
            alpha[v] = a;
            p0[v] = e0; p1[v] = e1;
        }

        // ----- C-layout -> A-layout via LDS bounce (bf16 P tile 16x32)
        __syncthreads();
#pragma unroll
        for (int v = 0; v < 8; ++v) {
            const int m = v + 8 * hi;
            lds_p[m * 32 + lanelo]      = (bf16)p0[v];
            lds_p[m * 32 + 16 + lanelo] = (bf16)p1[v];
        }
        __syncthreads();
        AFrag pf;
        pf.h[0] = *(const v8bf*)&lds_p[lanelo * 32 + alo];
        pf.h[1] = *(const v8bf*)&lds_p[lanelo * 32 + alo + 8];

        // ----- rescale accumulators, then ctx += P(16x32) @ V(32x64)
#pragma unroll
        for (int g = 0; g < 4; ++g)
#pragma unroll
            for (int v = 0; v < 8; ++v) acc[g][v] *= alpha[v];

#pragma unroll
        for (int g = 0; g < 4; ++g) {
            v16bf bv = *(const v16bf*)(lv + (g * 16 + lanelo) * 32 + klo);
            acc[g] = wmma_bf16(pf.v, bv, acc[g]);
        }
        buf ^= 1;
    }

    // ----- normalize and store ctx as [b, s, h*Dh + dh]
    const int b = bh / NHEAD, h = bh % NHEAD;
#pragma unroll
    for (int g = 0; g < 4; ++g) {
#pragma unroll
        for (int v = 0; v < 8; ++v) {
            const int s  = q0 + v + 8 * hi;
            const int dh = g * 16 + lanelo;
            const float val = acc[g][v] / lrow[v];
            ctx[((size_t)(b * S_LEN + s)) * DMODEL + h * DHEAD + dh] = (bf16)val;
        }
    }
}

// --------------------------------------------------------------- driver ----
extern "C" void kernel_launch(void* const* d_in, const int* in_sizes, int n_in,
                              void* d_out, int out_size, void* d_ws, size_t ws_size,
                              hipStream_t stream) {
    (void)in_sizes; (void)n_in; (void)out_size; (void)ws_size;
    const float* hs = (const float*)d_in[0];
    const float* Wq = (const float*)d_in[1];
    const float* bq = (const float*)d_in[2];
    const float* Wk = (const float*)d_in[3];
    const float* bk = (const float*)d_in[4];
    const float* Wv = (const float*)d_in[5];
    const float* bv = (const float*)d_in[6];
    const float* Wo = (const float*)d_in[7];
    const float* bo = (const float*)d_in[8];

    char* ws = (char*)d_ws;
    bf16* Xb   = (bf16*)(ws);                         //  8 MB  X bf16 [4096,1024]
    bf16* WqT  = (bf16*)(ws + 8388608);               //  2 MB each, pre-transposed [N,K]
    bf16* WkT  = (bf16*)(ws + 10485760);
    bf16* WvT  = (bf16*)(ws + 12582912);
    bf16* WoT  = (bf16*)(ws + 14680064);
    bf16* Qb   = (bf16*)(ws + 16777216);              //  8 MB  [bh, s, dh]
    bf16* Kb   = (bf16*)(ws + 25165824);              //  8 MB  [bh, s, dh]
    bf16* VTb  = (bf16*)(ws + 33554432);              //  8 MB  [bh, dh, s]
    bf16* ctxb = (bf16*)(ws + 41943040);              //  8 MB  [4096,1024]

    const int nX = MROWS * DMODEL;                    // 4,194,304
    cast_f32_bf16<<<(nX + 255) / 256, 256, 0, stream>>>(hs, Xb, nX);
    cast_transpose<<<4096, 256, 0, stream>>>(Wq, WqT);
    cast_transpose<<<4096, 256, 0, stream>>>(Wk, WkT);
    cast_transpose<<<4096, 256, 0, stream>>>(Wv, WvT);
    cast_transpose<<<4096, 256, 0, stream>>>(Wo, WoT);

    dim3 ggrid(DMODEL / 64, MROWS / 128);             // (16, 32)
    gemm_bf16<<<ggrid, 256, 0, stream>>>(Xb, WqT, bq, Qb, 0);
    gemm_bf16<<<ggrid, 256, 0, stream>>>(Xb, WkT, bk, Kb, 0);
    gemm_bf16<<<ggrid, 256, 0, stream>>>(Xb, WvT, bv, VTb, 2);

    flash_attn<<<2 * NHEAD * (S_LEN / 16), 32, 0, stream>>>(Qb, Kb, VTb, ctxb);

    gemm_bf16<<<ggrid, 256, 0, stream>>>(ctxb, WoT, bo, d_out, 3);
}
